// HierarchicalAttention_2765958939380
// MI455X (gfx1250) — matrix operand order repaired and checked by reading, not compiled, not run on hardware
//
#include <hip/hip_runtime.h>
#include <hip/hip_bf16.h>
#include <math.h>

// Problem constants
#define BB 2
#define SS 4096
#define DD 512
#define HH 8
#define DH 64

typedef __attribute__((ext_vector_type(16))) _Float16 v16h;
typedef __attribute__((ext_vector_type(8)))  _Float16 v8h;
typedef __attribute__((ext_vector_type(8)))  float    v8f;
typedef __attribute__((ext_vector_type(4)))  int      v4i;

#define WMMA_F16(a, b, c) \
  __builtin_amdgcn_wmma_f32_16x16x32_f16(false, (a), false, (b), (short)0, (c), false, false)

#ifndef __has_builtin
#define __has_builtin(x) 0
#endif

// -------- async global->LDS path (gfx1250 GLOBAL_LOAD_ASYNC_TO_LDS_B128) ----
// Probe-discovered signature: (int4 AS1*, int4 AS3*, imm offset, imm cpol)
#if __has_builtin(__builtin_amdgcn_global_load_async_to_lds_b128)
#define HAVE_ASYNC_LDS 1
#define AS1 __attribute__((address_space(1)))
#define AS3 __attribute__((address_space(3)))
#define ASYNC_CP16(gsrc, ldst) \
  __builtin_amdgcn_global_load_async_to_lds_b128((AS1 v4i*)(gsrc), (AS3 v4i*)(ldst), 0, 0)
#if __has_builtin(__builtin_amdgcn_s_wait_asynccnt)
#define WAIT_ASYNC(n) __builtin_amdgcn_s_wait_asynccnt(n)
#else
#define WAIT_ASYNC(n) asm volatile("s_wait_asynccnt %0" ::"i"(n) : "memory")
#endif
#else
#define HAVE_ASYNC_LDS 0
#endif

// ---------------------------------------------------------------------------
// Fragment loaders (CDNA5 wave32 WMMA layouts, ISA 7.12.2)
// A (16x32 f16, MxK): lane l holds row M=l%16; element i -> K = g*8 + (i<8 ? i : i+8), g=l/16
// B (32x16 f16, KxN): lane l holds col N=l%16; element i -> K = g*8 + (i<8 ? i : i+8), g=l/16
// C/D (16x16 f32):    VGPR j, lanes 0-15: (M=j,   N=lane); lanes 16-31: (M=j+8, N=lane-16)
// ---------------------------------------------------------------------------

__device__ __forceinline__ v16h load_a_f16(const _Float16* src, int ld, int r0, int kb, int lane) {
  const int g = (lane >> 4) & 1, m = lane & 15;
  const _Float16* p = src + (size_t)(r0 + m) * (size_t)ld + kb + g * 8;
  v8h lo = *(const v8h*)(p);
  v8h hi = *(const v8h*)(p + 16);
  v16h a;
#pragma unroll
  for (int i = 0; i < 8; ++i) { a[i] = lo[i]; a[8 + i] = hi[i]; }
  return a;
}

// src staged "N-major": element (K=k, N=n) at src[n*ld + k]
__device__ __forceinline__ v16h load_b_f16(const _Float16* src, int ld, int c0, int kb, int lane) {
  const int g = (lane >> 4) & 1, n = lane & 15;
  const _Float16* p = src + (size_t)(c0 + n) * (size_t)ld + kb + g * 8;
  v8h lo = *(const v8h*)(p);
  v8h hi = *(const v8h*)(p + 16);
  v16h b;
#pragma unroll
  for (int i = 0; i < 8; ++i) { b[i] = lo[i]; b[8 + i] = hi[i]; }
  return b;
}

// row reductions across a 16-lane group (wave32: xor masks <16 stay in group)
__device__ __forceinline__ float grpmax16(float v) {
#pragma unroll
  for (int m = 8; m >= 1; m >>= 1) v = fmaxf(v, __shfl_xor(v, m, 32));
  return v;
}
__device__ __forceinline__ float grpsum16(float v) {
#pragma unroll
  for (int m = 8; m >= 1; m >>= 1) v += __shfl_xor(v, m, 32);
  return v;
}

// ---------------------------------------------------------------------------
// Kernel 0a: convert + transpose weights to f16 W^T[n*512 + k] = W[k*512 + n]
// ---------------------------------------------------------------------------
__global__ __launch_bounds__(256) void prep_w(const float* __restrict__ Wq,
                                              const float* __restrict__ Wk,
                                              const float* __restrict__ Wv,
                                              const float* __restrict__ Wo,
                                              _Float16* __restrict__ Wt) {
  const int which = blockIdx.y;
  const float* src = (which == 0) ? Wq : (which == 1) ? Wk : (which == 2) ? Wv : Wo;
  const int idx = blockIdx.x * 256 + threadIdx.x;  // 0..262143
  const int n = idx >> 9, k = idx & 511;
  Wt[(size_t)which * (DD * DD) + idx] = (_Float16)src[(size_t)k * DD + n];
}

// ---------------------------------------------------------------------------
// Kernel 0b: convert x to f16 once
// ---------------------------------------------------------------------------
__global__ __launch_bounds__(256) void prep_x(const float* __restrict__ x,
                                              _Float16* __restrict__ xh) {
  const size_t i = ((size_t)blockIdx.x * 256 + threadIdx.x) * 8;
  float4 u0 = *(const float4*)(x + i);
  float4 u1 = *(const float4*)(x + i + 4);
  v8h o;
  o[0] = (_Float16)u0.x; o[1] = (_Float16)u0.y;
  o[2] = (_Float16)u0.z; o[3] = (_Float16)u0.w;
  o[4] = (_Float16)u1.x; o[5] = (_Float16)u1.y;
  o[6] = (_Float16)u1.z; o[7] = (_Float16)u1.w;
  *(v8h*)(xh + i) = o;
}

// ---------------------------------------------------------------------------
// Kernel 1: QKV projection GEMM.  Y = X*W + b  (f16 x f16 -> f32 acc -> f16)
// grid: (128, 32, 3), block 128 (4 waves). W tile staged once in LDS (async).
// Q,K stored [b,h,s,d];  V stored transposed [b,h,d,s]
// ---------------------------------------------------------------------------
__global__ __launch_bounds__(128) void qkv_gemm(const _Float16* __restrict__ xh,
                                                const _Float16* __restrict__ Wt,
                                                const float* __restrict__ bq,
                                                const float* __restrict__ bk,
                                                const float* __restrict__ bv,
                                                _Float16* __restrict__ Qh,
                                                _Float16* __restrict__ Kh,
                                                _Float16* __restrict__ Vt) {
  __shared__ alignas(16) _Float16 Wlds[16 * DD];  // 16 KB: 16 output cols x K=512

  const int tid = threadIdx.x;
  const int lane = tid & 31, wave = tid >> 5;
  const int rt = blockIdx.x * 4 + wave;   // 0..511  (rows of B*S)
  const int ct = blockIdx.y;              // 0..31   (cols of 512)
  const int which = blockIdx.z;           // 0=q 1=k 2=v
  const _Float16* W = Wt + (size_t)which * (DD * DD);
  const float* bias = (which == 0) ? bq : (which == 1) ? bk : bv;
  const int g = lane >> 4, n = lane & 15;
  const int r0 = rt * 16, c0 = ct * 16;

  // stage the W tile: 16 rows x 512 halves (64 x v8h per row)
#if HAVE_ASYNC_LDS
#pragma unroll
  for (int i = tid; i < 16 * 64; i += 128) {
    const int row = i >> 6, off = (i & 63) * 8;
    ASYNC_CP16(&W[(size_t)(c0 + row) * DD + off], &Wlds[row * DD + off]);
  }
  WAIT_ASYNC(0);
#else
#pragma unroll
  for (int i = tid; i < 16 * 64; i += 128) {
    const int row = i >> 6, off = (i & 63) * 8;
    *(v8h*)&Wlds[row * DD + off] = *(const v8h*)&W[(size_t)(c0 + row) * DD + off];
  }
#endif
  __syncthreads();

  v8f acc = {};
#pragma unroll 4
  for (int kb = 0; kb < DD; kb += 32) {
    v16h a = load_a_f16(xh, DD, r0, kb, lane);
    v16h b = load_b_f16(Wlds, DD, 0, kb, lane);   // ds_load path
    acc = WMMA_F16(a, b, acc);
  }

  const int c = c0 + n, h = c >> 6, d = c & 63;
  const float bval = bias[c];
#pragma unroll
  for (int j = 0; j < 8; ++j) {
    const int r = r0 + j + 8 * g;
    const int bi = r >> 12;          // r / 4096
    const int s = r & (SS - 1);
    const _Float16 v = (_Float16)(acc[j] + bval);
    if (which == 0)
      Qh[(((size_t)bi * HH + h) * SS + s) * DH + d] = v;
    else if (which == 1)
      Kh[(((size_t)bi * HH + h) * SS + s) * DH + d] = v;
    else
      Vt[(((size_t)bi * HH + h) * DH + d) * SS + s] = v;
  }
}

// ---------------------------------------------------------------------------
// Kernel 2: flash attention. 4 waves/block share one head; each wave owns a
// 16-query tile. K/V 32-key blocks double-buffered in LDS via async copies.
// grid: (64, B*H), block 128.
// ---------------------------------------------------------------------------
__global__ __launch_bounds__(128) void attn_fwd(const _Float16* __restrict__ Qh,
                                                const _Float16* __restrict__ Kh,
                                                const _Float16* __restrict__ Vt,
                                                _Float16* __restrict__ ctx) {
  __shared__ alignas(16) _Float16 Klds[2][32 * DH];   // 2 x 4 KB: keys x d
  __shared__ alignas(16) _Float16 Vlds[2][DH * 32];   // 2 x 4 KB: d x keys (V^T)
  __shared__ alignas(16) _Float16 Pbuf[4][16 * 32];

  const int tid = threadIdx.x;
  const int lane = tid & 31, wave = tid >> 5;
  const int bh = blockIdx.y;
  const int b = bh >> 3, h = bh & 7;
  const int qt = blockIdx.x * 4 + wave;   // 0..255
  const int s0 = qt * 16;
  const int g = lane >> 4, n = lane & 15;

  const size_t hoff = ((size_t)b * HH + h) * (size_t)SS * DH;
  const _Float16* Q = Qh + hoff;
  const _Float16* K = Kh + hoff;   // [s][d]
  const _Float16* V = Vt + hoff;   // [d][s]
  _Float16* P = &Pbuf[wave][0];

  const v16h qa0 = load_a_f16(Q, DH, s0, 0, lane);
  const v16h qa1 = load_a_f16(Q, DH, s0, 32, lane);

  v8f O0 = {}, O1 = {}, O2 = {}, O3 = {};
  float mrow[8], lrow[8];
#pragma unroll
  for (int j = 0; j < 8; ++j) { mrow[j] = -1e30f; lrow[j] = 0.0f; }
  const float scale = 0.125f;  // 1/sqrt(64)

  // one 32-key block: 4 score WMMAs, online softmax, P relayout, 4 PV WMMAs
  auto compute_block = [&](const _Float16* Kc, const _Float16* Vc) {
    v8f st0 = {}, st1 = {};
    v16h kf = load_b_f16(Kc, DH, 0, 0, lane);
    st0 = WMMA_F16(qa0, kf, st0);
    kf = load_b_f16(Kc, DH, 0, 32, lane);
    st0 = WMMA_F16(qa1, kf, st0);
    kf = load_b_f16(Kc, DH, 16, 0, lane);
    st1 = WMMA_F16(qa0, kf, st1);
    kf = load_b_f16(Kc, DH, 16, 32, lane);
    st1 = WMMA_F16(qa1, kf, st1);
#pragma unroll
    for (int j = 0; j < 8; ++j) {
      const float a0 = st0[j] * scale;
      const float a1 = st1[j] * scale;
      const float bmax = grpmax16(fmaxf(a0, a1));
      const float mnew = fmaxf(mrow[j], bmax);
      const float alpha = __expf(mrow[j] - mnew);
      const float p0 = __expf(a0 - mnew);
      const float p1 = __expf(a1 - mnew);
      lrow[j] = lrow[j] * alpha + grpsum16(p0 + p1);
      mrow[j] = mnew;
      O0[j] *= alpha; O1[j] *= alpha; O2[j] *= alpha; O3[j] *= alpha;
      const int m = j + 8 * g;
      P[m * 32 + n] = (_Float16)p0;
      P[m * 32 + 16 + n] = (_Float16)p1;
    }
    __builtin_amdgcn_wave_barrier();  // LDS store->load reuse within wave
    const v16h pa = load_a_f16(P, 32, 0, 0, lane);
    v16h vf = load_b_f16(Vc, 32, 0, 0, lane);
    O0 = WMMA_F16(pa, vf, O0);
    vf = load_b_f16(Vc, 32, 16, 0, lane);
    O1 = WMMA_F16(pa, vf, O1);
    vf = load_b_f16(Vc, 32, 32, 0, lane);
    O2 = WMMA_F16(pa, vf, O2);
    vf = load_b_f16(Vc, 32, 48, 0, lane);
    O3 = WMMA_F16(pa, vf, O3);
  };

#if HAVE_ASYNC_LDS
  // ---- double-buffered async K/V staging (ASYNCcnt-tracked LDS DMA) ----
  auto fill_async = [&](int buf, int kk) {
#pragma unroll
    for (int i = tid; i < 32 * 8; i += 128) {    // K: 32 rows x 8 v8h
      const int row = i >> 3, off = (i & 7) * 8;
      ASYNC_CP16(&K[(size_t)(kk + row) * DH + off], &Klds[buf][row * DH + off]);
    }
#pragma unroll
    for (int i = tid; i < DH * 4; i += 128) {    // V^T: 64 rows x 4 v8h
      const int row = i >> 2, off = (i & 3) * 8;
      ASYNC_CP16(&V[(size_t)row * SS + kk + off], &Vlds[buf][row * 32 + off]);
    }
  };
  fill_async(0, 0);
  int cur = 0;
  for (int kk = 0; kk < SS; kk += 32) {
    if (kk + 32 < SS) {
      fill_async(cur ^ 1, kk + 32);  // 4 new per-thread async copies in flight
      WAIT_ASYNC(4);                 // wait only for the current block's 4
    } else {
      WAIT_ASYNC(0);
    }
    __syncthreads();
    compute_block(&Klds[cur][0], &Vlds[cur][0]);
    __syncthreads();                 // all waves done before buffer reuse
    cur ^= 1;
  }
#else
  // ---- synchronous single-buffer fallback ----
  for (int kk = 0; kk < SS; kk += 32) {
#pragma unroll
    for (int i = tid; i < 32 * 8; i += 128) {
      const int row = i >> 3, off = (i & 7) * 8;
      *(v8h*)&Klds[0][row * DH + off] = *(const v8h*)&K[(size_t)(kk + row) * DH + off];
    }
#pragma unroll
    for (int i = tid; i < DH * 4; i += 128) {
      const int row = i >> 2, off = (i & 3) * 8;
      *(v8h*)&Vlds[0][row * 32 + off] = *(const v8h*)&V[(size_t)row * SS + kk + off];
    }
    if (kk + 32 < SS) {
      __builtin_prefetch(K + (size_t)(kk + 32) * DH, 0, 0);  // global_prefetch_b8
      __builtin_prefetch(V + (size_t)(kk + 32), 0, 0);
    }
    __syncthreads();
    compute_block(&Klds[0][0], &Vlds[0][0]);
    __syncthreads();
  }
#endif

  // ---- epilogue: O /= l, write ctx [b*S+s][h*64+d] as f16
#pragma unroll
  for (int j = 0; j < 8; ++j) {
    const float inv = 1.0f / lrow[j];
    const int m = j + 8 * g;
    const int s = s0 + m;
    const size_t rb = ((size_t)b * SS + s) * DD + h * DH;
    ctx[rb + 0 + n]  = (_Float16)(O0[j] * inv);
    ctx[rb + 16 + n] = (_Float16)(O1[j] * inv);
    ctx[rb + 32 + n] = (_Float16)(O2[j] * inv);
    ctx[rb + 48 + n] = (_Float16)(O3[j] * inv);
  }
}

// ---------------------------------------------------------------------------
// Kernel 3: output projection. out = ctx * Wo + bo  (f16 x f16 -> f32)
// grid: (128, 32), block 128 (4 waves). Wo tile staged in LDS (async).
// ---------------------------------------------------------------------------
__global__ __launch_bounds__(128) void out_proj(const _Float16* __restrict__ ctx,
                                                const _Float16* __restrict__ WtO,
                                                const float* __restrict__ bo,
                                                float* __restrict__ out) {
  __shared__ alignas(16) _Float16 Wlds[16 * DD];  // 16 KB

  const int tid = threadIdx.x;
  const int lane = tid & 31, wave = tid >> 5;
  const int rt = blockIdx.x * 4 + wave;
  const int ct = blockIdx.y;
  const int g = lane >> 4, n = lane & 15;
  const int r0 = rt * 16, c0 = ct * 16;

#if HAVE_ASYNC_LDS
#pragma unroll
  for (int i = tid; i < 16 * 64; i += 128) {
    const int row = i >> 6, off = (i & 63) * 8;
    ASYNC_CP16(&WtO[(size_t)(c0 + row) * DD + off], &Wlds[row * DD + off]);
  }
  WAIT_ASYNC(0);
#else
#pragma unroll
  for (int i = tid; i < 16 * 64; i += 128) {
    const int row = i >> 6, off = (i & 63) * 8;
    *(v8h*)&Wlds[row * DD + off] = *(const v8h*)&WtO[(size_t)(c0 + row) * DD + off];
  }
#endif
  __syncthreads();

  v8f acc = {};
#pragma unroll 4
  for (int kb = 0; kb < DD; kb += 32) {
    v16h a = load_a_f16(ctx, DD, r0, kb, lane);
    v16h b = load_b_f16(Wlds, DD, 0, kb, lane);
    acc = WMMA_F16(a, b, acc);
  }
  const int c = c0 + n;
  const float bval = bo[c];
#pragma unroll
  for (int j = 0; j < 8; ++j) {
    const int r = r0 + j + 8 * g;
    out[(size_t)r * DD + c] = acc[j] + bval;
  }
}

// ---------------------------------------------------------------------------
// Host launcher
// ---------------------------------------------------------------------------
extern "C" void kernel_launch(void* const* d_in, const int* in_sizes, int n_in,
                              void* d_out, int out_size, void* d_ws, size_t ws_size,
                              hipStream_t stream) {
  (void)in_sizes; (void)n_in; (void)out_size; (void)ws_size;
  const float* x  = (const float*)d_in[0];
  const float* Wq = (const float*)d_in[1];
  const float* bq = (const float*)d_in[2];
  const float* Wk = (const float*)d_in[3];
  const float* bk = (const float*)d_in[4];
  const float* Wv = (const float*)d_in[5];
  const float* bv = (const float*)d_in[6];
  const float* Wo = (const float*)d_in[7];
  const float* bo = (const float*)d_in[8];
  float* out = (float*)d_out;

  // workspace layout (bytes): Wt (2 MiB) | xh (8) | Qh (8) | Kh (8) | Vt (8) | ctx (8)
  char* ws = (char*)d_ws;
  _Float16* Wt  = (_Float16*)(ws);
  _Float16* xh  = (_Float16*)(ws + (size_t)(2u << 20));
  _Float16* Qh  = (_Float16*)(ws + (size_t)(10u << 20));
  _Float16* Kh  = (_Float16*)(ws + (size_t)(18u << 20));
  _Float16* Vt  = (_Float16*)(ws + (size_t)(26u << 20));
  _Float16* ctx = (_Float16*)(ws + (size_t)(34u << 20));
  _Float16* WtO = Wt + (size_t)3 * DD * DD;

  hipLaunchKernelGGL(prep_w, dim3(1024, 4), dim3(256), 0, stream, Wq, Wk, Wv, Wo, Wt);
  hipLaunchKernelGGL(prep_x, dim3(2048), dim3(256), 0, stream, x, xh);
  hipLaunchKernelGGL(qkv_gemm, dim3(128, 32, 3), dim3(128), 0, stream,
                     xh, Wt, bq, bk, bv, Qh, Kh, Vt);
  hipLaunchKernelGGL(attn_fwd, dim3(64, 16), dim3(128), 0, stream, Qh, Kh, Vt, ctx);
  hipLaunchKernelGGL(out_proj, dim3(128, 32), dim3(128), 0, stream, ctx, WtO, bo, out);
}
